// OuterProductMean_91207925498265
// MI455X (gfx1250) — matrix operand order repaired and checked by reading, not compiled, not run on hardware
//
#include <hip/hip_runtime.h>

#define S_MSA 512
#define N_TOK 384
#define C_MSA 64
#define C_HID 32
#define C_PAIR 128
#define M_DIM (N_TOK * C_HID) /* 12288 */

typedef __attribute__((ext_vector_type(16))) __bf16 v16bf;
typedef __attribute__((ext_vector_type(8)))  float  v8f;

union Frag {
    unsigned int u[8];
    v16bf v;
};

__device__ __forceinline__ unsigned short f2bf(float f) {
    union { float f; unsigned int u; } x;
    x.f = f;
    unsigned int r = x.u + 0x7FFFu + ((x.u >> 16) & 1u);
    return (unsigned short)(r >> 16);
}

// Low 32 bits of a generic pointer into __shared__ = LDS byte offset on amdgcn.
__device__ __forceinline__ unsigned lds_off(const void* p) {
    return (unsigned)(unsigned long long)p;
}

// ---------------------------------------------------------------------------
// Kernel 1: LayerNorm + a/b projections, store bf16 panels in [ic][s] layout
// (k-major so WMMA K-pairs are contiguous 32-bit LDS reads later).
// One wave per (s,i) row. 8 waves / block.
// ---------------------------------------------------------------------------
__global__ __launch_bounds__(256)
void opm_ln_proj(const float* __restrict__ msa, const float* __restrict__ gamma,
                 const float* __restrict__ beta, const float* __restrict__ w_a,
                 const float* __restrict__ w_b, unsigned short* __restrict__ Abf,
                 unsigned short* __restrict__ Bbf) {
    __shared__ float was[C_HID][C_MSA + 1];   // +1 pad: conflict-free column reads
    __shared__ float wbs[C_HID][C_MSA + 1];
    __shared__ float mrow[8][C_MSA];

    const int t = threadIdx.x;
    const int w = t >> 5;
    const int lane = t & 31;

    for (int idx = t; idx < C_HID * C_MSA; idx += 256) {
        int rr = idx >> 6, kk = idx & 63;
        was[rr][kk] = w_a[idx];
        wbs[rr][kk] = w_b[idx];
    }
    __syncthreads();

    const int row = blockIdx.x * 8 + w;       // row over (s,i); grid sized exactly
    const int s = row / N_TOK;
    const int i = row - s * N_TOK;
    const float* xr = msa + (size_t)row * C_MSA;

    float x0 = xr[2 * lane + 0];
    float x1 = xr[2 * lane + 1];

    float sum = x0 + x1;
    #pragma unroll
    for (int off = 16; off >= 1; off >>= 1) sum += __shfl_xor(sum, off, 32);
    const float mean = sum * (1.0f / C_MSA);
    const float d0 = x0 - mean, d1 = x1 - mean;
    float sq = d0 * d0 + d1 * d1;
    #pragma unroll
    for (int off = 16; off >= 1; off >>= 1) sq += __shfl_xor(sq, off, 32);
    const float rinv = rsqrtf(sq * (1.0f / C_MSA) + 1e-5f);

    mrow[w][2 * lane + 0] = d0 * rinv * gamma[2 * lane + 0] + beta[2 * lane + 0];
    mrow[w][2 * lane + 1] = d1 * rinv * gamma[2 * lane + 1] + beta[2 * lane + 1];
    __syncthreads();

    float aa = 0.f, bb = 0.f;
    #pragma unroll
    for (int k = 0; k < C_MSA; ++k) {
        float mv = mrow[w][k];                 // LDS broadcast across lanes
        aa += mv * was[lane][k];
        bb += mv * wbs[lane][k];
    }
    const size_t o = (size_t)(i * C_HID + lane) * S_MSA + s;   // [ic][s]
    Abf[o] = f2bf(aa);
    Bbf[o] = f2bf(bb);
}

// ---------------------------------------------------------------------------
// Kernel 2: w_out -> bf16 (row-major [p][cd], cd contiguous = WMMA B K-pairs)
// ---------------------------------------------------------------------------
__global__ __launch_bounds__(256)
void opm_cvt_wout(const float* __restrict__ w_out, unsigned short* __restrict__ Wbf) {
    const int idx = blockIdx.x * 256 + threadIdx.x;
    Wbf[idx] = f2bf(w_out[idx]);
}

// ---------------------------------------------------------------------------
// Kernel 3: fused  o = (A^T B)/S   then   z = o @ w_out^T + b_out
// Workgroup: 128 x 128 o-tile = 4 i's x 32 c  by  4 j's x 32 d.
// 8 waves (wave32): wave grid 2x4, each wave 4x2 subtiles of 16x16.
// K staged with double-buffered GLOBAL_LOAD_ASYNC_TO_LDS (ASYNCcnt) so the
// DMA of chunk c+1 overlaps the v_wmma work of chunk c.
// Phase 2 reuses the LDS (union) for the o tile.
// ---------------------------------------------------------------------------
__global__ __launch_bounds__(256)
void opm_main(const unsigned short* __restrict__ Abf,
              const unsigned short* __restrict__ Bbf,
              const unsigned short* __restrict__ Wbf,
              const float* __restrict__ b_out,
              float* __restrict__ z) {
    __shared__ union {
        struct { unsigned short A[2][128][40]; unsigned short B[2][128][40]; } p1; // 40960 B
        unsigned short o[16][1040];                                                // 33280 B
    } sm;

    const int t    = threadIdx.x;
    const int w    = t >> 5;        // wave 0..7
    const int lane = t & 31;
    const int r    = lane & 15;     // lane row/col within fragment
    const int h    = lane >> 4;     // lane half (K-half select per ISA layout)
    const int waveM = w >> 2;       // 0..1  -> 64 rows each
    const int waveN = w & 3;        // 0..3  -> 32 cols each

    const v8f vzero = {};
    v8f acc[4][2];
    #pragma unroll
    for (int mt = 0; mt < 4; ++mt)
        #pragma unroll
        for (int nt = 0; nt < 2; ++nt) acc[mt][nt] = vzero;

    const size_t gMa = (size_t)blockIdx.y * 128;
    const size_t gNb = (size_t)blockIdx.x * 128;

    // Async-stage one K=32 chunk (128 rows x 64 B) of both panels into LDS buf.
    // 256 threads x 2 passes x 16 B cover each 8 KB panel; ASYNCcnt += 4/wave.
    auto stage = [&](int buf, int k0) {
        #pragma unroll
        for (int pass = 0; pass < 2; ++pass) {
            const int idx = t + pass * 256;   // 0..511
            const int rr = idx >> 2;          // row 0..127
            const int sg = idx & 3;           // 16B segment 0..3
            const unsigned short* ga = Abf + (gMa + rr) * S_MSA + k0 + sg * 8;
            const unsigned short* gb = Bbf + (gNb + rr) * S_MSA + k0 + sg * 8;
            const unsigned la = lds_off(&sm.p1.A[buf][rr][sg * 8]);
            const unsigned lb = lds_off(&sm.p1.B[buf][rr][sg * 8]);
            asm volatile("global_load_async_to_lds_b128 %0, %1, off\n\t"
                         "global_load_async_to_lds_b128 %2, %3, off"
                         :: "v"(la), "v"(ga), "v"(lb), "v"(gb)
                         : "memory");
        }
    };

    stage(0, 0);
    for (int c = 0; c < S_MSA / 32; ++c) {
        const int cur = c & 1;
        // Drain our async batch, then block-wide barrier: chunk c visible to all
        // waves AND every wave is done reading the buffer we overwrite next.
        asm volatile("s_wait_asynccnt 0x0" ::: "memory");
        __syncthreads();
        if (c + 1 < S_MSA / 32) stage(cur ^ 1, (c + 1) * 32);

        Frag bF[2];
        #pragma unroll
        for (int nt = 0; nt < 2; ++nt) {
            const int rowN = waveN * 32 + nt * 16 + r;
            #pragma unroll
            for (int v = 0; v < 8; ++v) {       // B: K split per ISA table
                const int k = (v < 4) ? (8 * h + 2 * v) : (16 + 8 * h + 2 * (v - 4));
                bF[nt].u[v] = *reinterpret_cast<const unsigned int*>(&sm.p1.B[cur][rowN][k]);
            }
        }
        #pragma unroll
        for (int mt = 0; mt < 4; ++mt) {
            Frag aF;
            const int rowM = waveM * 64 + mt * 16 + r;
            #pragma unroll
            for (int v = 0; v < 8; ++v) {       // A: K split per ISA table
                const int k = (v < 4) ? (8 * h + 2 * v)
                                      : (16 + 8 * h + 2 * (v - 4));
                aF.u[v] = *reinterpret_cast<const unsigned int*>(&sm.p1.A[cur][rowM][k]);
            }
            #pragma unroll
            for (int nt = 0; nt < 2; ++nt) {
                acc[mt][nt] = __builtin_amdgcn_wmma_f32_16x16x32_bf16(
                    false, aF.v, false, bF[nt].v, (short)0, acc[mt][nt], false, false);
            }
        }
    }
    __syncthreads();   // all waves done with p1 before the union is repurposed

    // ---- scatter o tile (scaled 1/S) to LDS as bf16, reindexed [ij][c*32+d]
    const float invS = 1.0f / (float)S_MSA;
    #pragma unroll
    for (int mt = 0; mt < 4; ++mt) {
        #pragma unroll
        for (int nt = 0; nt < 2; ++nt) {
            #pragma unroll
            for (int v = 0; v < 8; ++v) {         // C/D layout: M = v + 8h, N = r
                const int Ml = waveM * 64 + mt * 16 + v + 8 * h;
                const int Nl = waveN * 32 + nt * 16 + r;
                const int ij = (Ml >> 5) * 4 + (Nl >> 5);
                const int cd = (Ml & 31) * 32 + (Nl & 31);
                sm.o[ij][cd] = f2bf(acc[mt][nt][v] * invS);
            }
        }
    }
    __syncthreads();

    // ---- phase 2: z[ij,p] = sum_cd o[ij,cd] * w_out[p,cd]  (M=16,N=128,K=1024)
    v8f acc2 = vzero;
    const int pw = w * 16;                        // each wave owns 16 p-columns
    for (int k0 = 0; k0 < C_HID * C_HID; k0 += 32) {
        Frag aF, bF;
        #pragma unroll
        for (int v = 0; v < 8; ++v) {
            const int ka = (v < 4) ? (k0 + 8 * h + 2 * v)
                                   : (k0 + 16 + 8 * h + 2 * (v - 4));
            aF.u[v] = *reinterpret_cast<const unsigned int*>(&sm.o[r][ka]);
            const int kb = ka;
            bF.u[v] = *reinterpret_cast<const unsigned int*>(
                &Wbf[(size_t)(pw + r) * (C_HID * C_HID) + kb]);
        }
        acc2 = __builtin_amdgcn_wmma_f32_16x16x32_bf16(
            false, aF.v, false, bF.v, (short)0, acc2, false, false);
    }

    const int p = pw + r;
    const float bo = b_out[p];
    #pragma unroll
    for (int v = 0; v < 8; ++v) {
        const int ij = v + 8 * h;                 // D row = (i_local, j_local)
        const int ig = blockIdx.y * 4 + (ij >> 2);
        const int jg = blockIdx.x * 4 + (ij & 3);
        z[((size_t)ig * N_TOK + jg) * C_PAIR + p] = acc2[v] + bo;
    }
}

// ---------------------------------------------------------------------------
extern "C" void kernel_launch(void* const* d_in, const int* in_sizes, int n_in,
                              void* d_out, int out_size, void* d_ws, size_t ws_size,
                              hipStream_t stream) {
    (void)in_sizes; (void)n_in; (void)out_size; (void)ws_size;
    const float* msa   = (const float*)d_in[0];
    const float* gamma = (const float*)d_in[1];
    const float* beta  = (const float*)d_in[2];
    const float* w_a   = (const float*)d_in[3];
    const float* w_b   = (const float*)d_in[4];
    const float* w_out = (const float*)d_in[5];
    const float* b_out = (const float*)d_in[6];
    float* z = (float*)d_out;

    unsigned short* Abf = (unsigned short*)d_ws;                    // 12.6 MB
    unsigned short* Bbf = Abf + (size_t)M_DIM * S_MSA;              // 12.6 MB
    unsigned short* Wbf = Bbf + (size_t)M_DIM * S_MSA;              // 256 KB

    opm_ln_proj<<<dim3(S_MSA * N_TOK / 8), dim3(256), 0, stream>>>(
        msa, gamma, beta, w_a, w_b, Abf, Bbf);
    opm_cvt_wout<<<dim3((C_PAIR * C_HID * C_HID) / 256), dim3(256), 0, stream>>>(
        w_out, Wbf);
    opm_main<<<dim3(N_TOK / 4, N_TOK / 4), dim3(256), 0, stream>>>(
        Abf, Bbf, Wbf, b_out, z);
}
